// MoeMLP_16655883174266
// MI455X (gfx1250) — matrix operand order repaired and checked by reading, not compiled, not run on hardware
//
#include <hip/hip_runtime.h>
#include <hip/hip_bf16.h>
#include <stdint.h>

// MI455X / gfx1250: wave32, WMMA bf16 16x16x32 (f32 accum), async global->LDS.

typedef __attribute__((ext_vector_type(8)))  float  v8f;
typedef __attribute__((ext_vector_type(4)))  __bf16 v4bf;
typedef __attribute__((ext_vector_type(8)))  __bf16 v8bf;
typedef __attribute__((ext_vector_type(16))) __bf16 v16bf;
typedef __attribute__((ext_vector_type(4)))  int    v4i;

#define E_    8
#define D_    1024
#define DFFN  2048
#define T_    8192
#define TK_   16384        // T*K
#define CAP_  16896        // T*K + E*64
#define NBLK  (CAP_/64)    // 264 row-blocks of 64
#define NCH   256          // chunks of 64 items over TK_
#define NKI   (D_/32)      // 32 K-iterations of 32

// ---------------- router: one wave32 per token, top-2 of logits ----------------
__global__ __launch_bounds__(256) void k_router(const float* __restrict__ x,
                                                const float* __restrict__ wr,
                                                int* __restrict__ sel)
{
    const int wave = threadIdx.x >> 5;
    const int lane = threadIdx.x & 31;
    const int t = blockIdx.x * 8 + wave;
    const float* xr = x + (size_t)t * D_;
    float acc[E_];
#pragma unroll
    for (int e = 0; e < E_; ++e) acc[e] = 0.f;
    for (int i = 0; i < D_ / 32; ++i) {
        const int d = i * 32 + lane;
        const float xv = xr[d];
#pragma unroll
        for (int e = 0; e < E_; ++e) acc[e] += xv * wr[e * D_ + d];
    }
#pragma unroll
    for (int e = 0; e < E_; ++e) {
#pragma unroll
        for (int m = 16; m >= 1; m >>= 1) acc[e] += __shfl_xor(acc[e], m, 32);
    }
    if (lane == 0) {
        // softmax is monotone; topv unused downstream -> top-2 of logits,
        // stable ties (first index wins) to match jax.lax.top_k
        float v1 = -3.4e38f; int e1 = 0;
#pragma unroll
        for (int e = 0; e < E_; ++e) if (acc[e] > v1) { v1 = acc[e]; e1 = e; }
        float v2 = -3.4e38f; int e2 = 0;
#pragma unroll
        for (int e = 0; e < E_; ++e) if (e != e1 && acc[e] > v2) { v2 = acc[e]; e2 = e; }
        sel[2 * t]     = e1;
        sel[2 * t + 1] = e2;
    }
}

// ---------------- one-time f32 -> bf16 of x (+64 zero pad rows) ---------------
__global__ __launch_bounds__(256) void k_cvt_x(const float* __restrict__ x,
                                               __bf16* __restrict__ xc)
{
    const size_t i = ((size_t)blockIdx.x * 256 + threadIdx.x) * 8;
    v8bf o;
    if (i < (size_t)T_ * D_) {
        const float4 f0 = *(const float4*)(x + i);
        const float4 f1 = *(const float4*)(x + i + 4);
        o[0] = (__bf16)f0.x; o[1] = (__bf16)f0.y; o[2] = (__bf16)f0.z; o[3] = (__bf16)f0.w;
        o[4] = (__bf16)f1.x; o[5] = (__bf16)f1.y; o[6] = (__bf16)f1.z; o[7] = (__bf16)f1.w;
    } else {
#pragma unroll
        for (int ii = 0; ii < 8; ++ii) o[ii] = (__bf16)0.f;   // zero pad rows
    }
    *(v8bf*)(xc + i) = o;
}

// ---------------- one-time f32 -> bf16 transpose: w1t[col][k] -----------------
__global__ __launch_bounds__(256) void k_cvt_w1t(const float* __restrict__ w1,
                                                 __bf16* __restrict__ w1t)
{
    __shared__ __bf16 t[64][72];             // 64k x 64c tile, padded rows
    const int tid = threadIdx.x;
    const int k0 = (blockIdx.x & 15) * 64;
    const int c0 = (blockIdx.x >> 4) * 64;
#pragma unroll
    for (int j = 0; j < 4; ++j) {
        const int q = tid + j * 256;         // 64 k-rows x 16 col-groups of 4
        const int k = q >> 4, cg = q & 15;
        const float4 f = *(const float4*)(w1 + (size_t)(k0 + k) * (E_ * DFFN) + c0 + cg * 4);
        v4bf v;
        v[0] = (__bf16)f.x; v[1] = (__bf16)f.y; v[2] = (__bf16)f.z; v[3] = (__bf16)f.w;
        *(v4bf*)&t[k][cg * 4] = v;
    }
    __syncthreads();
#pragma unroll
    for (int j = 0; j < 4; ++j) {
        const int q = tid + j * 256;         // 64 c-rows x 4 k-groups of 16
        const int c = q >> 2, kg = q & 3;
        v8bf o0, o1;
#pragma unroll
        for (int ii = 0; ii < 8; ++ii) {
            o0[ii] = t[kg * 16 + ii][c];
            o1[ii] = t[kg * 16 + 8 + ii][c];
        }
        __bf16* op = w1t + (size_t)(c0 + c) * D_ + k0 + kg * 16;
        *(v8bf*)op       = o0;
        *(v8bf*)(op + 8) = o1;
    }
}

// ---------------- chunked per-expert histogram (deterministic, no atomics) ----
__global__ void k_chunk_count(const int* __restrict__ sel, int* __restrict__ cc)
{
    const int c = threadIdx.x;   // 256 chunks x 64 items
    int cnt[E_];
#pragma unroll
    for (int e = 0; e < E_; ++e) cnt[e] = 0;
    for (int i = 0; i < 64; ++i) {
        const int ee = sel[c * 64 + i];
#pragma unroll
        for (int e = 0; e < E_; ++e) cnt[e] += (ee == e) ? 1 : 0;
    }
#pragma unroll
    for (int e = 0; e < E_; ++e) cc[c * E_ + e] = cnt[e];
}

// ---------------- exclusive scan over chunks + padded offsets + block experts -
__global__ void k_scan(const int* __restrict__ cc, int* __restrict__ cbase,
                       int* __restrict__ cum_p, int* __restrict__ expert_of)
{
    __shared__ int tot[E_];
    const int e = threadIdx.x;
    if (e < E_) {
        int run = 0;
        for (int c = 0; c < NCH; ++c) { cbase[c * E_ + e] = run; run += cc[c * E_ + e]; }
        tot[e] = run;
    }
    __syncthreads();
    if (threadIdx.x == 0) {
        int cp = 0; cum_p[0] = 0;
        for (int i = 0; i < E_; ++i) {
            const int g = (tot[i] + 63) & ~63;   // pad to BLOCK=64
            cp += g; cum_p[i + 1] = cp;
        }
        for (int b = 0; b < NBLK; ++b) {
            const int row = b * 64; int eo = 0;
            for (int i = 0; i < E_; ++i) if (row >= cum_p[i] && row < cum_p[i + 1]) eo = i;
            expert_of[b] = eo;   // rows past cum_p[E]: expert 0, all src=-1 -> zeros
        }
    }
}

__global__ void k_init_src(int* __restrict__ src)
{
    const int i = blockIdx.x * 256 + threadIdx.x;
    if (i < CAP_) src[i] = -1;
}

// ---------------- stable scatter: src_row[dst] = token id ---------------------
__global__ void k_scatter(const int* __restrict__ sel, const int* __restrict__ cbase,
                          const int* __restrict__ cum_p, int* __restrict__ src)
{
    const int c = threadIdx.x;   // one chunk per thread, serial within chunk = stable
    int run[E_], base[E_];
#pragma unroll
    for (int e = 0; e < E_; ++e) { run[e] = cbase[c * E_ + e]; base[e] = cum_p[e]; }
    for (int i = 0; i < 64; ++i) {
        const int idx = c * 64 + i;
        const int ee  = sel[idx];
        int dst = 0;
#pragma unroll
        for (int e = 0; e < E_; ++e) {
            if (ee == e) dst = base[e] + run[e];
            run[e] += (ee == e) ? 1 : 0;
        }
        src[dst] = idx >> 1;     // order // K, K=2
    }
}

// ---------------- grouped GEMM: A gathered from global bf16, B async->LDS -----
__global__ __launch_bounds__(256) void k_moe_gemm(
    const __bf16* __restrict__ xc, const __bf16* __restrict__ w1t,
    const int* __restrict__ src, const int* __restrict__ expert_of,
    float* __restrict__ out)
{
    // B tile [128 n][48 k-stride] bf16, double buffered (16B-aligned rows)
    __shared__ __bf16 Bb[2][128 * 48];

    const int tid = threadIdx.x;
    const int rm = blockIdx.x * 64;     // 64-row tiles never straddle experts
    const int cn = blockIdx.y * 128;
    const int e  = expert_of[blockIdx.x];

    const int wave = tid >> 5, lane = tid & 31;
    const int wm = (wave & 3) * 16;     // wave's M sub-tile
    const int wn = (wave >> 2) * 64;    // wave's N range (4 x 16)
    const int n16 = lane & 15, kh = lane >> 4;

    // Per-lane A row (K-invariant): padding rows (-1) hit the zeroed pad row T_.
    const int sr = src[rm + wm + n16];
    const size_t arow = (size_t)((sr < 0) ? T_ : sr);
    const __bf16* Abase = xc + arow * D_ + kh * 8;

    const __bf16* Bgbl = w1t + ((size_t)e * DFFN + cn) * D_;   // row stride D_

    auto issueB = [&](int buf, int kt) {
        const int k0 = kt * 32;
#pragma unroll
        for (int j = 0; j < 2; ++j) {
            const int q = tid + j * 256;       // 128 n-rows x 4 k-groups of 8
            const int n = q >> 2, kg = q & 3;
            const __bf16* gp = Bgbl + (size_t)n * D_ + k0 + kg * 8;
            __bf16* lp = &Bb[buf][n * 48 + kg * 8];
#if __has_builtin(__builtin_amdgcn_global_load_async_to_lds_b128)
            __builtin_amdgcn_global_load_async_to_lds_b128(
                (__attribute__((address_space(1))) v4i*)(uintptr_t)gp,
                (__attribute__((address_space(3))) v4i*)(uintptr_t)lp,
                0, 0);
#else
            *(v8bf*)lp = *(const v8bf*)gp;     // sync fallback
#endif
        }
    };

    v8f acc[4];
#pragma unroll
    for (int s = 0; s < 4; ++s)
#pragma unroll
        for (int v = 0; v < 8; ++v) acc[s][v] = 0.f;

    issueB(0, 0);

    for (int kt = 0; kt < NKI; ++kt) {
        const int buf = kt & 1;
#if __has_builtin(__builtin_amdgcn_s_wait_asynccnt)
        __builtin_amdgcn_s_wait_asynccnt(0);
#else
        asm volatile("s_wait_asynccnt 0" ::: "memory");
#endif
        __syncthreads();                        // all waves' B[buf] visible
        if (kt + 1 < NKI) issueB(buf ^ 1, kt + 1);

        // A fragment straight from global bf16 (16-bit A 16x32 per-lane layout)
        const __bf16* Ap = Abase + kt * 32;
        const v8bf a0 = *(const v8bf*)Ap;        // K kh*8 .. kh*8+7
        const v8bf a1 = *(const v8bf*)(Ap + 16); // K 16+kh*8 ..
        if (kt + 4 < NKI) __builtin_prefetch(Ap + 4 * 32, 0, 1);
        const v16bf a = __builtin_shufflevector(a0, a1,
            0, 1, 2, 3, 4, 5, 6, 7, 8, 9, 10, 11, 12, 13, 14, 15);
#pragma unroll
        for (int s = 0; s < 4; ++s) {
            // B fragment: lane n = col, K kh*8..kh*8+7 and 16+kh*8..16+kh*8+7 in LDS [n][k]
            const __bf16* Bp = &Bb[buf][(wn + s * 16 + n16) * 48 + kh * 8];
            const v8bf b0 = *(const v8bf*)Bp;
            const v8bf b1 = *(const v8bf*)(Bp + 16);
            const v16bf b = __builtin_shufflevector(b0, b1,
                0, 1, 2, 3, 4, 5, 6, 7, 8, 9, 10, 11, 12, 13, 14, 15);
            acc[s] = __builtin_amdgcn_wmma_f32_16x16x32_bf16(
                false, a, false, b, (short)0, acc[s], false, false);
        }
    }

    // C/D layout: lane n = N col, VGPR v -> M = v + (lane>=16 ? 8 : 0)
#pragma unroll
    for (int s = 0; s < 4; ++s) {
        const int col = cn + wn + s * 16 + n16;
        float* op = out + (size_t)(rm + wm + kh * 8) * DFFN + col;
#pragma unroll
        for (int v = 0; v < 8; ++v) op[(size_t)v * DFFN] = acc[s][v];
    }
}

extern "C" void kernel_launch(void* const* d_in, const int* in_sizes, int n_in,
                              void* d_out, int out_size, void* d_ws, size_t ws_size,
                              hipStream_t stream)
{
    const float* x  = (const float*)d_in[0];   // [T, D]
    const float* wr = (const float*)d_in[1];   // [E, D]
    const float* w1 = (const float*)d_in[2];   // [D, E*DFFN]
    float* out = (float*)d_out;                // [CAP, DFFN]

    uint8_t* ws = (uint8_t*)d_ws;
    int* sel       = (int*)ws;       // 16384
    int* cc        = sel + 16384;    // 2048
    int* cbase     = cc + 2048;      // 2048
    int* cum_p     = cbase + 2048;   // 16 (padded)
    int* expert_of = cum_p + 16;     // 272 (padded)
    int* src       = expert_of + 272;// 16896  -> ends ~150.7 KB

    const size_t XC_OFF  = 262144;                         // 256 KB aligned
    const size_t XC_BYTES = (size_t)(T_ + 64) * D_ * 2;    // 16,908,288
    __bf16* xc  = (__bf16*)(ws + XC_OFF);
    __bf16* w1t = (__bf16*)(ws + XC_OFF + XC_BYTES);       // 33,554,432 B

    k_cvt_x  <<<(T_ + 64) * (D_ / 8) / 256, 256, 0, stream>>>(x, xc);
    k_cvt_w1t<<<16 * (E_ * DFFN / 64),      256, 0, stream>>>(w1, w1t);

    k_router<<<T_ / 8, 256, 0, stream>>>(x, wr, sel);
    k_chunk_count<<<1, 256, 0, stream>>>(sel, cc);
    k_scan<<<1, 64, 0, stream>>>(cc, cbase, cum_p, expert_of);
    k_init_src<<<(CAP_ + 255) / 256, 256, 0, stream>>>(src);
    k_scatter<<<1, 256, 0, stream>>>(sel, cbase, cum_p, src);

    k_moe_gemm<<<dim3(NBLK, DFFN / 128), 256, 0, stream>>>(xc, w1t, src, expert_of, out);
}
